// HelicalCell_17841294147990
// MI455X (gfx1250) — matrix operand order repaired and checked by reading, not compiled, not run on hardware
//
#include <hip/hip_runtime.h>
#include <hip/hip_bf16.h>

// ---------------------------------------------------------------------------
// HelicalCell for MI455X (gfx1250, wave32, WMMA).
//   B=32, T=1024, D=H=1024.  outs[B,T,H] f32 + coh_sum scalar.
//
//   k0: zero sync/stats + h0
//   k1: pack wx_w / wy_w / mix_w into f16 WMMA-B tile-major layout (L2 resident)
//   k2: Yall[b*T+t, :] = e @ wy_w + wy_b  (data-parallel WMMA GEMM, f16 out)
//   k3: persistent cooperative recurrence: 32 WGs x 128 thr (4 waves),
//       WG j owns H-columns [32j,32j+32) = 4 16x16 WMMA tiles (M=32).
//       3 grid barriers per step (atomic generation barrier).
//   All WMMA K-loops use dual accumulators (break D->C dependency chain) and
//   global_prefetch of upcoming L2-resident weight tiles.
// ---------------------------------------------------------------------------

typedef __attribute__((ext_vector_type(16))) _Float16 v16h;
typedef __attribute__((ext_vector_type(8)))  _Float16 v8h;
typedef __attribute__((ext_vector_type(8)))  float    v8f;

#define B_   32
#define T_   1024
#define D_   1024
#define H_   1024
#define K3_  3072
#define NWG  32
#define LAM_ 0.01f
#define LN_EPS_ 1e-5f
#define KSTR ((H_ / 16) * 512)   // packed halves per K-tile row (all weights have N=1024)

// ---- workspace layout (bytes) ----
#define SYNC_OFF  0                                   // cnt, gen
#define STAT_OFF  256                                 // 5x32 floats + coh accum
#define H32_OFF   4096                                // h state f32   (128KB)
#define H16_OFF   (H32_OFF + B_*H_*4)                 // h state f16   (64KB)
#define ACT_OFF   (H16_OFF + B_*H_*2)                 // [b|a|c] f16   (192KB)
#define WXP_OFF   (ACT_OFF + B_*K3_*2)                // wx packed f16 (2MB)
#define WYP_OFF   (WXP_OFF + (H_/32)*(H_/16)*512*2)   // wy packed f16 (2MB)
#define MIXP_OFF  (WYP_OFF + (D_/32)*(H_/16)*512*2)   // mix packed    (6MB)
#define YALL_OFF  (MIXP_OFF + (K3_/32)*(H_/16)*512*2) // Yall f16      (64MB)
// total ~78MB (weights + Yall are L2-friendly; h/act/stats stay L2-hot)

__device__ inline v8f wmma_f16(v16h a, v16h b, v8f c) {
  // emits v_wmma_f32_16x16x32_f16
  return __builtin_amdgcn_wmma_f32_16x16x32_f16(
      /*neg_a=*/false, a, /*neg_b=*/false, b,
      /*c_mod=*/(short)0, c, /*reuse_a=*/false, /*reuse_b=*/false);
}

// A operand (16x32 f16, row-major source, row = lane&15):
//   lane<16 holds K {kk..kk+7, kk+16..kk+23}; lane>=16 the +8 shifted set.
//   Caller passes p = rowptr + kk + ((lane&16)?8:0); two 16B loads.
__device__ inline v16h load_a16(const _Float16* p) {
  v8h lo = *(const v8h*)(p);
  v8h hi = *(const v8h*)(p + 16);
  v16h a;
#pragma unroll
  for (int i = 0; i < 8; ++i) { a[i] = lo[i]; a[i + 8] = hi[i]; }
  return a;
}

__device__ inline float redux16(float v) {  // sum over each 16-lane half
  v += __shfl_xor(v, 1);
  v += __shfl_xor(v, 2);
  v += __shfl_xor(v, 4);
  v += __shfl_xor(v, 8);
  return v;
}

// generation-counter grid barrier (all NWG blocks guaranteed resident)
__device__ inline void gbar(unsigned* cnt, unsigned* gen, unsigned nwg) {
  __syncthreads();
  if (threadIdx.x == 0) {
    __threadfence();
    unsigned g = __hip_atomic_load(gen, __ATOMIC_ACQUIRE, __HIP_MEMORY_SCOPE_AGENT);
    unsigned a = __hip_atomic_fetch_add(cnt, 1u, __ATOMIC_ACQ_REL, __HIP_MEMORY_SCOPE_AGENT) + 1u;
    if (a == nwg) {
      __hip_atomic_store(cnt, 0u, __ATOMIC_RELAXED, __HIP_MEMORY_SCOPE_AGENT);
      __hip_atomic_fetch_add(gen, 1u, __ATOMIC_RELEASE, __HIP_MEMORY_SCOPE_AGENT);
    } else {
      while (__hip_atomic_load(gen, __ATOMIC_ACQUIRE, __HIP_MEMORY_SCOPE_AGENT) == g)
        __builtin_amdgcn_s_sleep(2);
    }
    __threadfence();
  }
  __syncthreads();
}

__device__ inline float gelu_exact(float x) {
  return 0.5f * x * (1.0f + erff(x * 0.70710678118654752f));
}

// ---------------------------------------------------------------------------
__global__ void init_ws_kernel(unsigned* w, int nwords) {
  int i = blockIdx.x * blockDim.x + threadIdx.x;
  if (i < nwords) w[i] = 0u;
}

// Pack row-major f32 W[K,N] into f16 WMMA B-operand tile order:
// tile (kt,nt) -> 512 halves; element (lane,i): k = kt*32 + (i&7) + (lane>=16?8:0) + (i>=8?16:0),
// n = nt*16 + (lane&15).  B load in GEMM = one contiguous 32B v16h per lane.
__global__ void pack_b16_kernel(const float* __restrict__ w, _Float16* __restrict__ dst,
                                int K, int N) {
  int gid = blockIdx.x * blockDim.x + threadIdx.x;
  int total = (K / 32) * (N / 16) * 512;
  if (gid >= total) return;
  int e    = gid & 511;
  int tile = gid >> 9;
  int ntiles = N / 16;
  int nt = tile % ntiles, kt = tile / ntiles;
  int lane = e >> 4, i = e & 15;
  int k = kt * 32 + (i & 7) + ((lane & 16) ? 8 : 0) + ((i & 8) ? 16 : 0);
  int n = nt * 16 + (lane & 15);
  dst[gid] = (_Float16)w[(size_t)k * N + n];
}

// Yall[R, :] = e[R, :] @ wy + wy_b, R = rb*T + t.  1 tile/wave, 8 waves/block.
__global__ __launch_bounds__(256)
void ygemm_kernel(const float* __restrict__ e, const float* __restrict__ wy_b,
                  const _Float16* __restrict__ wyp, _Float16* __restrict__ yall) {
  const int lane = threadIdx.x & 31, wid = threadIdx.x >> 5;
  const int tileId = blockIdx.x * 8 + wid;
  const int rt = tileId >> 6;        // row tile (B*T/16 = 2048)
  const int ct = tileId & 63;        // col tile (H/16 = 64)
  const int aoff = (lane & 16) ? 8 : 0;
  const float* arow = e + (size_t)(rt * 16 + (lane & 15)) * D_ + aoff;
  const _Float16* bp = wyp + (size_t)ct * 512 + lane * 16;
  v8f c0 = {}, c1 = {};
  for (int kk = 0; kk < D_; kk += 64) {
    __builtin_prefetch(arow + kk + 128, 0, 0);                   // e rows, 2 iters ahead
    __builtin_prefetch(bp + (size_t)((kk >> 5) + 4) * KSTR, 0, 0); // weight tiles ahead
    v8f f0 = *(const v8f*)(arow + kk);
    v8f f1 = *(const v8f*)(arow + kk + 16);
    v8f f2 = *(const v8f*)(arow + kk + 32);
    v8f f3 = *(const v8f*)(arow + kk + 48);
    v16h a0, a1;
#pragma unroll
    for (int i = 0; i < 8; ++i) {
      a0[i] = (_Float16)f0[i]; a0[i + 8] = (_Float16)f1[i];
      a1[i] = (_Float16)f2[i]; a1[i + 8] = (_Float16)f3[i];
    }
    v16h b0 = *(const v16h*)(bp + (size_t)(kk >> 5) * KSTR);
    v16h b1 = *(const v16h*)(bp + (size_t)((kk >> 5) + 1) * KSTR);
    c0 = wmma_f16(a0, b0, c0);
    c1 = wmma_f16(a1, b1, c1);
  }
  const int col = ct * 16 + (lane & 15);
  const int rowadd = (lane & 16) ? 8 : 0;
  const float bias = wy_b[col];
#pragma unroll
  for (int r = 0; r < 8; ++r) {
    int row = rt * 16 + r + rowadd;
    yall[(size_t)row * H_ + col] = (_Float16)(c0[r] + c1[r] + bias);
  }
}

// ---------------------------------------------------------------------------
__global__ __launch_bounds__(128)
void helical_rnn_kernel(char* ws,
                        const float* __restrict__ wx_b, const float* __restrict__ mix_b,
                        const float* __restrict__ ln_g, const float* __restrict__ ln_b,
                        const int* __restrict__ use_coh, float* __restrict__ out) {
  float*    hf32 = (float*)(ws + H32_OFF);
  _Float16* hf16 = (_Float16*)(ws + H16_OFF);
  _Float16* act  = (_Float16*)(ws + ACT_OFF);
  const _Float16* wxp  = (const _Float16*)(ws + WXP_OFF);
  const _Float16* mixp = (const _Float16*)(ws + MIXP_OFF);
  const _Float16* yall = (const _Float16*)(ws + YALL_OFF);
  unsigned* cnt = (unsigned*)(ws + SYNC_OFF);
  unsigned* gen = cnt + 1;
  float* ssum = (float*)(ws + STAT_OFF);
  float* ssq  = ssum + 32;
  float* cnum = ssum + 64;
  float* cnp  = ssum + 96;
  float* cnn  = ssum + 128;
  float* cacc = ssum + 160;

  const int wg   = blockIdx.x;
  const int lane = threadIdx.x & 31;
  const int wid  = threadIdx.x >> 5;          // 4 waves: (mt, nt)
  const int mt = wid & 1, nt = wid >> 1;
  const int rowbase = mt * 16;
  const int colbase = wg * 32 + nt * 16;
  const int col  = colbase + (lane & 15);
  const int rowadd = (lane & 16) ? 8 : 0;
  const int aoff   = (lane & 16) ? 8 : 0;
  const int arow   = rowbase + (lane & 15);   // A-operand source row
  const int coh_on = use_coh[0];

  const _Float16* wxp_l  = wxp  + (size_t)(colbase >> 4) * 512 + lane * 16;
  const _Float16* mixp_l = mixp + (size_t)(colbase >> 4) * 512 + lane * 16;

  // cos/sin of [5,7,11,13]*2pi/24 (cycled by t%4)
  const float CT[4] = { 0.25881904510252074f, -0.25881904510252074f,
                       -0.96592582628906829f, -0.96592582628906829f};
  const float ST[4] = { 0.96592582628906829f,  0.96592582628906829f,
                        0.25881904510252074f, -0.25881904510252074f};

  for (int t = 0; t < T_; ++t) {
    // --- WG0: fold previous step's coherence, then zero the stats slots ---
    if (wg == 0 && wid == 0) {
      if (lane == 0 && t > 0 && coh_on) {
        float s = 0.f;
        for (int rb = 0; rb < B_; ++rb) {
          float den = fmaxf(sqrtf(cnp[rb]), 1e-8f) * fmaxf(sqrtf(cnn[rb]), 1e-8f);
          s += cnum[rb] / den;
        }
        *cacc += LAM_ * (1.0f - s * (1.0f / B_));
      }
      ssum[lane] = 0.f; ssq[lane] = 0.f;
      cnum[lane] = 0.f; cnp[lane] = 0.f; cnn[lane] = 0.f;
    }

    // --- phase 1: X = h @ wx + wx_b ; emit [b|a|c] slice in f16 ---
    v8f c0 = {}, c1 = {};
    {
      const _Float16* ha = hf16 + (size_t)arow * H_ + aoff;
      for (int kk = 0; kk < H_; kk += 64) {
        __builtin_prefetch(wxp_l + (size_t)((kk >> 5) + 4) * KSTR, 0, 0);
        v16h a0 = load_a16(ha + kk);
        v16h b0 = *(const v16h*)(wxp_l + (size_t)(kk >> 5) * KSTR);
        c0 = wmma_f16(a0, b0, c0);
        v16h a1 = load_a16(ha + kk + 32);
        v16h b1 = *(const v16h*)(wxp_l + (size_t)((kk >> 5) + 1) * KSTR);
        c1 = wmma_f16(a1, b1, c1);
      }
    }
#pragma unroll
    for (int r = 0; r < 8; ++r) {
      int row = rowbase + r + rowadd;
      float X = c0[r] + c1[r] + wx_b[col];
      float Y = (float)yall[((size_t)row * T_ + t) * H_ + col];
      float bb = 0.5f * (Y - X);
      float aa = sqrtf((fabsf(X) + 1e-6f) * (fabsf(Y) + 1e-6f)); // exp(0.5(log+log))
      float cc = 0.5f * (Y + X);
      size_t ab = (size_t)row * K3_;
      act[ab + col]          = (_Float16)bb;
      act[ab + H_ + col]     = (_Float16)aa;
      act[ab + 2 * H_ + col] = (_Float16)cc;
    }
    gbar(cnt, gen, NWG);

    // --- phase 2: z = act @ mix + mix_b ; u = gelu(z) + 0.1*rot(h_prev) ---
    v8f z0 = {}, z1 = {};
    {
      const _Float16* aa2 = act + (size_t)arow * K3_ + aoff;
      for (int kk = 0; kk < K3_; kk += 64) {
        __builtin_prefetch(mixp_l + (size_t)((kk >> 5) + 4) * KSTR, 0, 0);
        v16h a0 = load_a16(aa2 + kk);
        v16h b0 = *(const v16h*)(mixp_l + (size_t)(kk >> 5) * KSTR);
        z0 = wmma_f16(a0, b0, z0);
        v16h a1 = load_a16(aa2 + kk + 32);
        v16h b1 = *(const v16h*)(mixp_l + (size_t)((kk >> 5) + 1) * KSTR);
        z1 = wmma_f16(a1, b1, z1);
      }
    }
    float u[8];
    const float cs = CT[t & 3], sn = ST[t & 3];
#pragma unroll
    for (int r = 0; r < 8; ++r) {
      int row = rowbase + r + rowadd;
      float z  = gelu_exact(z0[r] + z1[r] + mix_b[col]);
      float x0 = hf32[(size_t)row * H_ + (col & ~1)];
      float y0 = hf32[(size_t)row * H_ + (col | 1)];
      float rot = (col & 1) ? (sn * x0 + cs * y0) : (cs * x0 - sn * y0);
      u[r] = z + 0.1f * rot;
      float s1 = redux16(u[r]);
      float s2 = redux16(u[r] * u[r]);
      if ((lane & 15) == 0) { atomicAdd(&ssum[row], s1); atomicAdd(&ssq[row], s2); }
    }
    gbar(cnt, gen, NWG);

    // --- phase 3: LayerNorm + GELU -> h_new ; outputs + coherence partials ---
#pragma unroll
    for (int r = 0; r < 8; ++r) {
      int row = rowbase + r + rowadd;
      float m    = ssum[row] * (1.0f / H_);
      float var  = ssq[row] * (1.0f / H_) - m * m;
      float rstd = rsqrtf(var + LN_EPS_);
      float yv = (u[r] - m) * rstd * ln_g[col] + ln_b[col];
      float hn = gelu_exact(yv);
      float hp = hf32[(size_t)row * H_ + col];
      out[((size_t)row * T_ + t) * H_ + col] = hn;     // outs[B,T,H]
      hf32[(size_t)row * H_ + col] = hn;
      hf16[(size_t)row * H_ + col] = (_Float16)hn;
      if (coh_on) {
        float n1 = redux16(hp * hn);
        float n2 = redux16(hp * hp);
        float n3 = redux16(hn * hn);
        if ((lane & 15) == 0) {
          atomicAdd(&cnum[row], n1); atomicAdd(&cnp[row], n2); atomicAdd(&cnn[row], n3);
        }
      }
    }
    gbar(cnt, gen, NWG);
  }

  // fold the final step's coherence and emit the scalar
  if (wg == 0 && threadIdx.x == 0) {
    float acc = *cacc;
    if (coh_on) {
      float s = 0.f;
      for (int rb = 0; rb < B_; ++rb) {
        float den = fmaxf(sqrtf(cnp[rb]), 1e-8f) * fmaxf(sqrtf(cnn[rb]), 1e-8f);
        s += cnum[rb] / den;
      }
      acc += LAM_ * (1.0f - s * (1.0f / B_));
    }
    out[(size_t)B_ * T_ * H_] = acc;
  }
}

// ---------------------------------------------------------------------------
extern "C" void kernel_launch(void* const* d_in, const int* in_sizes, int n_in,
                              void* d_out, int out_size, void* d_ws, size_t ws_size,
                              hipStream_t stream) {
  const float* e     = (const float*)d_in[0];
  const float* wx_w  = (const float*)d_in[1];
  const float* wx_b  = (const float*)d_in[2];
  const float* wy_w  = (const float*)d_in[3];
  const float* wy_b  = (const float*)d_in[4];
  const float* mix_w = (const float*)d_in[5];
  const float* mix_b = (const float*)d_in[6];
  const float* ln_g  = (const float*)d_in[7];
  const float* ln_b  = (const float*)d_in[8];
  const int*   ucoh  = (const int*)d_in[9];
  char*  ws  = (char*)d_ws;
  float* out = (float*)d_out;

  // k0: zero sync + stats + h state (ws is poisoned 0xAA by the harness)
  {
    int nwords = ACT_OFF / 4;
    hipLaunchKernelGGL(init_ws_kernel, dim3((nwords + 255) / 256), dim3(256), 0, stream,
                       (unsigned*)ws, nwords);
  }
  // k1: pack weights to f16 WMMA-B tile order (stays resident in 192MB L2)
  {
    int tot = (H_ / 32) * (H_ / 16) * 512;
    hipLaunchKernelGGL(pack_b16_kernel, dim3((tot + 255) / 256), dim3(256), 0, stream,
                       wx_w, (_Float16*)(ws + WXP_OFF), H_, H_);
    hipLaunchKernelGGL(pack_b16_kernel, dim3((tot + 255) / 256), dim3(256), 0, stream,
                       wy_w, (_Float16*)(ws + WYP_OFF), D_, H_);
    int totm = (K3_ / 32) * (H_ / 16) * 512;
    hipLaunchKernelGGL(pack_b16_kernel, dim3((totm + 255) / 256), dim3(256), 0, stream,
                       mix_w, (_Float16*)(ws + MIXP_OFF), K3_, H_);
  }
  // k2: hoisted data-parallel GEMM Yall = e @ wy + wy_b (68.7 GFLOP)
  {
    int blocks = (B_ * T_ / 16) * (H_ / 16) / 8;  // 1 WMMA tile per wave
    hipLaunchKernelGGL(ygemm_kernel, dim3(blocks), dim3(256), 0, stream,
                       e, wy_b, (const _Float16*)(ws + WYP_OFF), (_Float16*)(ws + YALL_OFF));
  }
  // k3: persistent cooperative recurrence (32 WGs, 3 grid barriers / step)
  hipLaunchKernelGGL(helical_rnn_kernel, dim3(NWG), dim3(128), 0, stream,
                     ws, wx_b, mix_b, ln_g, ln_b, ucoh, out);
}
